// STConvEncoder_45973329936766
// MI455X (gfx1250) — matrix operand order repaired and checked by reading, not compiled, not run on hardware
//
#include <hip/hip_runtime.h>
#include <hip/hip_bf16.h>

#define N_NODES 10000
#define T_STEPS 12
#define B_BATCH 8
#define BT 96
#define C_IN 2
#define H_DIM 64
#define E_EDGES 320000
#define ROWS_PER_BLOCK 128   // 8 waves * 16 rows

typedef _Float16 v16h __attribute__((ext_vector_type(16)));
typedef _Float16 h8   __attribute__((ext_vector_type(8)));
typedef float    v8f  __attribute__((ext_vector_type(8)));

static __device__ inline v8f wmma32f16(v16h a, v16h b, v8f c) {
  // (neg_a, A, neg_b, B, c_mod, C, reuse_a, reuse_b)
  return __builtin_amdgcn_wmma_f32_16x16x32_f16(false, a, false, b, (short)0, c,
                                                false, false);
}

// convert 8 f32 to 8 f16, zeroed when !ok (branchless)
static __device__ inline h8 cvt8(float4 a, float4 b, bool ok) {
  h8 h;
  h[0] = (_Float16)(ok ? a.x : 0.0f);
  h[1] = (_Float16)(ok ? a.y : 0.0f);
  h[2] = (_Float16)(ok ? a.z : 0.0f);
  h[3] = (_Float16)(ok ? a.w : 0.0f);
  h[4] = (_Float16)(ok ? b.x : 0.0f);
  h[5] = (_Float16)(ok ? b.y : 0.0f);
  h[6] = (_Float16)(ok ? b.z : 0.0f);
  h[7] = (_Float16)(ok ? b.w : 0.0f);
  return h;
}

// ---------------------------------------------------------------- utilities
__global__ void fill_f32(float* p, float v, long n) {
  long i = (long)blockIdx.x * blockDim.x + threadIdx.x;
  if (i < n) p[i] = v;
}

__global__ void negcopy_f32(float* dst, const float* src, long n) {
  long i = (long)blockIdx.x * blockDim.x + threadIdx.x;
  if (i < n) dst[i] = -src[i];
}

// ------------------------------------------------------- graph normalization
__global__ void deg_kernel(const int* ei, const float* ew, float* deg) {
  int e = blockIdx.x * blockDim.x + threadIdx.x;
  if (e < E_EDGES) atomicAdd(&deg[ei[e]], ew[e]);
}

__global__ void dis_kernel(const float* deg, float* dis) {
  int n = blockIdx.x * blockDim.x + threadIdx.x;
  if (n < N_NODES) {
    float d = deg[n];
    dis[n] = (d > 0.0f) ? rsqrtf(d) : 0.0f;
  }
}

__global__ void what_kernel(const int* ei, const float* ew, const float* dis,
                            float* w_hat) {
  int e = blockIdx.x * blockDim.x + threadIdx.x;
  if (e < E_EDGES) {
    int row = ei[e];
    int col = ei[E_EDGES + e];
    w_hat[e] = -dis[row] * ew[e] * dis[col];
  }
}

// ---- pre-swizzle Cheb weights [3][64][64] into B-fragment order (f16)
// B-frag (32x16): lane = n + 16*(kw>>4); slot = kw & 15
__global__ void swizzle_cheb_w(const float* __restrict__ W, _Float16* __restrict__ out) {
  int i = blockIdx.x * blockDim.x + threadIdx.x;
  if (i >= 3 * H_DIM * H_DIM) return;
  int k   = i / (H_DIM * H_DIM);
  int rem = i % (H_DIM * H_DIM);
  int ki  = rem / H_DIM;          // K row
  int j   = rem % H_DIM;          // output col
  int ks = ki >> 5, kw = ki & 31;
  int nt = j >> 4,  n  = j & 15;
  int lane = n + ((kw >> 4) << 4);
  int slot = kw & 15;
  out[(((k * 2 + ks) * 4 + nt) * 32 + lane) * 16 + slot] = (_Float16)W[i];
}

// ---- pre-swizzle tconv2 weights [O=64][C=64][3] (im2col K = dt*64+c) -> f16 frags
__global__ void swizzle_tc2_w(const float* __restrict__ w1, const float* __restrict__ w2,
                              const float* __restrict__ w3, _Float16* __restrict__ out) {
  int i = blockIdx.x * blockDim.x + threadIdx.x;
  if (i >= 3 * 192 * H_DIM) return;
  int g   = i / (192 * H_DIM);
  int rem = i % (192 * H_DIM);
  int ki  = rem / H_DIM;          // K = dt*64 + c
  int o   = rem % H_DIM;
  const float* w = (g == 0) ? w1 : (g == 1) ? w2 : w3;
  int dt = ki >> 6, c = ki & 63;
  float v = w[(o * H_DIM + c) * 3 + dt];
  int ks = ki >> 5, kw = ki & 31;
  int nt = o >> 4,  n  = o & 15;
  int lane = n + ((kw >> 4) << 4);
  int slot = kw & 15;
  out[(((g * 6 + ks) * 4 + nt) * 32 + lane) * 16 + slot] = (_Float16)v;
}

// ------------------------------------------------------------- temporal conv 1
__global__ void tconv1_kernel(const float* __restrict__ X,
                              const float* __restrict__ w1, const float* __restrict__ b1,
                              const float* __restrict__ w2, const float* __restrict__ b2,
                              const float* __restrict__ w3, const float* __restrict__ b3,
                              float* __restrict__ T0) {
  long idx = (long)blockIdx.x * blockDim.x + threadIdx.x;
  if (idx >= (long)BT * N_NODES * H_DIM) return;
  int h  = (int)(idx & (H_DIM - 1));
  long nh = idx >> 6;
  int n  = (int)(nh % N_NODES);
  int bt = (int)(nh / N_NODES);
  int t = bt % T_STEPS, b = bt / T_STEPS;

  float p = b1[h], q = b2[h], r = b3[h];
#pragma unroll
  for (int dt = 0; dt < 3; ++dt) {
    int tt = t + dt - 1;
    if (tt < 0 || tt >= T_STEPS) continue;
    const float* xp = X + (((long)(b * T_STEPS + tt) * N_NODES + n) * C_IN);
#pragma unroll
    for (int c = 0; c < C_IN; ++c) {
      float xv = xp[c];
      int wi = (h * C_IN + c) * 3 + dt;
      p = fmaf(xv, w1[wi], p);
      q = fmaf(xv, w2[wi], q);
      r = fmaf(xv, w3[wi], r);
    }
  }
  float sig = 1.0f / (1.0f + __expf(-q));
  T0[idx] = fmaxf(fmaf(p, sig, r), 0.0f);
}

// ------------------------------------------------------ sparse Chebyshev prop
__global__ void cheb_prop_kernel(const float* __restrict__ src,
                                 float* __restrict__ dst,
                                 const int* __restrict__ ei,
                                 const float* __restrict__ w_hat, float scale) {
  int warp = (blockIdx.x * blockDim.x + threadIdx.x) >> 5;
  int lane = threadIdx.x & 31;
  if (warp >= E_EDGES) return;
  int row = ei[warp];
  int col = ei[E_EDGES + warp];
  float w = w_hat[warp] * scale;
  const long SLICE = (long)N_NODES * H_DIM;
  const float* sp = src + (long)col * H_DIM + lane * 2;
  float*       dp = dst + (long)row * H_DIM + lane * 2;
#pragma unroll 1
  for (int bt = 0; bt < BT; ++bt) {
    long o = (long)bt * SLICE;
    if (bt + 1 < BT) __builtin_prefetch(sp + o + SLICE, 0, 0);  // global_prefetch_b8
    float2 v = *(const float2*)(sp + o);
    atomicAdd(dp + o,     v.x * w);
    atomicAdd(dp + o + 1, v.y * w);
  }
}

// --------------------------------------------- ChebConv GEMM + bias + ReLU
__global__ void __launch_bounds__(256)
cheb_gemm_kernel(const float* __restrict__ Tx0, const float* __restrict__ Tx1,
                 const float* __restrict__ Tx2, const _Float16* __restrict__ wf,
                 const float* __restrict__ bias, float* __restrict__ S) {
  // all three A tiles in fragment order: [k(3)][wave(8)][ks(2)][lane(32)][slot(16)]
  __shared__ __attribute__((aligned(32))) _Float16 laA[3 * 8 * 2 * 32 * 16]; // 48 KB

  int bt   = blockIdx.y;
  int row0 = blockIdx.x * ROWS_PER_BLOCK;
  int tid  = threadIdx.x;
  int wave = tid >> 5, lane = tid & 31;

  long off = ((long)bt * N_NODES + row0) * H_DIM;
  const float* bases[3] = {Tx0 + off, Tx1 + off, Tx2 + off};

  // branchless staging: 8 channels/thread/step, 2x b128 load -> 1x b128 LDS store
#pragma unroll
  for (int it = 0; it < 12; ++it) {
    int idx = it * 256 + tid;       // [0, 3*1024)
    int k   = it >> 2;              // tile id (1024 chunks per tile)
    int pt  = idx & 1023;
    int r   = pt >> 3;              // row in tile
    int co  = (pt & 7) << 3;        // channel start (multiple of 8)
    bool ok = (row0 + r) < N_NODES;
    const float* p = bases[k] + (long)(ok ? r : 0) * H_DIM + co;
    float4 va = *(const float4*)p;
    float4 vb = *(const float4*)(p + 4);
    h8 hv = cvt8(va, vb, ok);
    int ks = co >> 5, kw = co & 31;
    int lane_s = (r & 15) + ((kw >> 4) << 4);
    int slot0  = kw & 15;           // 0 or 8 -> 16B aligned
    *(h8*)(laA + ((((k * 8 + (r >> 4)) * 2) + ks) * 32 + lane_s) * 16 + slot0) = hv;
  }
  __syncthreads();

  v8f acc[4] = {};
#pragma unroll
  for (int k = 0; k < 3; ++k) {
#pragma unroll
    for (int ks = 0; ks < 2; ++ks) {
      v16h af = *(const v16h*)(laA + ((((k * 8 + wave) * 2) + ks) * 32 + lane) * 16);
#pragma unroll
      for (int nt = 0; nt < 4; ++nt) {
        v16h bf = *(const v16h*)(wf + (((k * 2 + ks) * 4 + nt) * 32 + lane) * 16);
        acc[nt] = wmma32f16(af, bf, acc[nt]);
      }
    }
  }

  int n0 = lane & 15;
  int mb = (lane >> 4) * 8;
#pragma unroll
  for (int nt = 0; nt < 4; ++nt) {
    int col = nt * 16 + n0;
    float bv = bias[col];
#pragma unroll
    for (int r = 0; r < 8; ++r) {
      int grow = row0 + wave * 16 + mb + r;
      if (grow < N_NODES)
        S[((long)bt * N_NODES + grow) * H_DIM + col] = fmaxf(acc[nt][r] + bv, 0.0f);
    }
  }
}

// ------------------------------------------------------------- temporal conv 2
__global__ void __launch_bounds__(256)
tconv2_kernel(const float* __restrict__ S, const _Float16* __restrict__ wf,
              const float* __restrict__ b1, const float* __restrict__ b2,
              const float* __restrict__ b3, float* __restrict__ out) {
  // A tile (128 x 192 im2col) in fragment order: [wave][ks(6)][lane][slot] = 48 KB
  __shared__ __attribute__((aligned(32))) _Float16 laA[8 * 6 * 32 * 16];

  int bt   = blockIdx.y;
  int b    = bt / T_STEPS, t = bt % T_STEPS;
  int row0 = blockIdx.x * ROWS_PER_BLOCK;
  int tid  = threadIdx.x;
  int wave = tid >> 5, lane = tid & 31;

  // branchless staging: 8 K-values/thread/step (same dt within a chunk)
#pragma unroll
  for (int it = 0; it < 12; ++it) {
    int idx = it * 256 + tid;       // [0, 3072)
    int r   = idx / 24;             // row in tile (24 chunks per row)
    int kc  = (idx % 24) << 3;      // K start (multiple of 8)
    int dt  = kc >> 6, c = kc & 63;
    int tt  = t + dt - 1;
    int grow = row0 + r;
    bool ok = (tt >= 0) & (tt < T_STEPS) & (grow < N_NODES);
    int ttc   = min(max(tt, 0), T_STEPS - 1);
    int growc = min(grow, N_NODES - 1);
    const float* p = S + ((((long)b * T_STEPS + ttc) * N_NODES + growc) * H_DIM + c);
    float4 va = *(const float4*)p;
    float4 vb = *(const float4*)(p + 4);
    h8 hv = cvt8(va, vb, ok);
    int ks = kc >> 5, kw = kc & 31;
    int lane_s = (r & 15) + ((kw >> 4) << 4);
    int slot0  = kw & 15;
    *(h8*)(laA + (((r >> 4) * 6 + ks) * 32 + lane_s) * 16 + slot0) = hv;
  }
  __syncthreads();

  v8f accP[4] = {}, accQ[4] = {}, accR[4] = {};

#pragma unroll
  for (int ks = 0; ks < 6; ++ks) {
    v16h af = *(const v16h*)(laA + ((wave * 6 + ks) * 32 + lane) * 16);
#pragma unroll
    for (int nt = 0; nt < 4; ++nt) {
      v16h bfP = *(const v16h*)(wf + (((0 * 6 + ks) * 4 + nt) * 32 + lane) * 16);
      accP[nt] = wmma32f16(af, bfP, accP[nt]);
      v16h bfQ = *(const v16h*)(wf + (((1 * 6 + ks) * 4 + nt) * 32 + lane) * 16);
      accQ[nt] = wmma32f16(af, bfQ, accQ[nt]);
      v16h bfR = *(const v16h*)(wf + (((2 * 6 + ks) * 4 + nt) * 32 + lane) * 16);
      accR[nt] = wmma32f16(af, bfR, accR[nt]);
    }
  }

  int n0 = lane & 15;
  int mb = (lane >> 4) * 8;
#pragma unroll
  for (int nt = 0; nt < 4; ++nt) {
    int o = nt * 16 + n0;
    float bp = b1[o], bq = b2[o], br = b3[o];
#pragma unroll
    for (int r = 0; r < 8; ++r) {
      int grow = row0 + wave * 16 + mb + r;
      if (grow >= N_NODES) continue;
      float p = accP[nt][r] + bp;
      float q = accQ[nt][r] + bq;
      float rr = accR[nt][r] + br;
      float sig = 1.0f / (1.0f + __expf(-q));
      out[((long)bt * N_NODES + grow) * H_DIM + o] = fmaxf(fmaf(p, sig, rr), 0.0f);
    }
  }
}

// ---------------------------------------------------------------------- host
extern "C" void kernel_launch(void* const* d_in, const int* in_sizes, int n_in,
                              void* d_out, int out_size, void* d_ws, size_t ws_size,
                              hipStream_t stream) {
  const float* X    = (const float*)d_in[0];
  const int*   ei   = (const int*)d_in[1];
  const float* ew   = (const float*)d_in[2];
  const float* t1w1 = (const float*)d_in[3];
  const float* t1b1 = (const float*)d_in[4];
  const float* t1w2 = (const float*)d_in[5];
  const float* t1b2 = (const float*)d_in[6];
  const float* t1w3 = (const float*)d_in[7];
  const float* t1b3 = (const float*)d_in[8];
  const float* chW  = (const float*)d_in[9];
  const float* chb  = (const float*)d_in[10];
  const float* t2w1 = (const float*)d_in[11];
  const float* t2b1 = (const float*)d_in[12];
  const float* t2w2 = (const float*)d_in[13];
  const float* t2b2 = (const float*)d_in[14];
  const float* t2w3 = (const float*)d_in[15];
  const float* t2b3 = (const float*)d_in[16];

  const long SL = (long)BT * N_NODES * H_DIM;   // 61,440,000 floats per tensor
  float* T0    = (float*)d_ws;
  float* Tx1   = T0 + SL;
  float* Tx2   = Tx1 + SL;
  float* Sb    = Tx2 + SL;
  float* deg   = Sb + SL;
  float* dis   = deg + N_NODES;
  float* w_hat = dis + N_NODES;
  _Float16* chWf = (_Float16*)(w_hat + E_EDGES);          // 12,288 halves
  _Float16* t2Wf = chWf + 3 * 2 * 4 * 32 * 16;            // 36,864 halves

  dim3 blk(256);

  // graph normalization: deg -> dis -> w_hat
  fill_f32<<<(N_NODES + 255) / 256, blk, 0, stream>>>(deg, 0.0f, N_NODES);
  deg_kernel<<<(E_EDGES + 255) / 256, blk, 0, stream>>>(ei, ew, deg);
  dis_kernel<<<(N_NODES + 255) / 256, blk, 0, stream>>>(deg, dis);
  what_kernel<<<(E_EDGES + 255) / 256, blk, 0, stream>>>(ei, ew, dis, w_hat);

  // pre-swizzle weights into WMMA B-fragment order (f16)
  swizzle_cheb_w<<<(3 * H_DIM * H_DIM + 255) / 256, blk, 0, stream>>>(chW, chWf);
  swizzle_tc2_w<<<(3 * 192 * H_DIM + 255) / 256, blk, 0, stream>>>(t2w1, t2w2, t2w3, t2Wf);

  // temporal conv 1 -> T0 (= Tx0)
  tconv1_kernel<<<(unsigned)((SL + 255) / 256), blk, 0, stream>>>(
      X, t1w1, t1b1, t1w2, t1b2, t1w3, t1b3, T0);

  // Tx1 = L_hat @ Tx0
  fill_f32<<<(unsigned)((SL + 255) / 256), blk, 0, stream>>>(Tx1, 0.0f, SL);
  cheb_prop_kernel<<<E_EDGES / 8, blk, 0, stream>>>(T0, Tx1, ei, w_hat, 1.0f);

  // Tx2 = 2 * L_hat @ Tx1 - Tx0
  negcopy_f32<<<(unsigned)((SL + 255) / 256), blk, 0, stream>>>(Tx2, T0, SL);
  cheb_prop_kernel<<<E_EDGES / 8, blk, 0, stream>>>(Tx1, Tx2, ei, w_hat, 2.0f);

  // S = relu(sum_k Tx_k @ W_k + b)   [WMMA]
  dim3 ggrid((N_NODES + ROWS_PER_BLOCK - 1) / ROWS_PER_BLOCK, BT);
  cheb_gemm_kernel<<<ggrid, blk, 0, stream>>>(T0, Tx1, Tx2, chWf, chb, Sb);

  // temporal conv 2 -> d_out   [WMMA, im2col K=192]
  tconv2_kernel<<<ggrid, blk, 0, stream>>>(Sb, t2Wf, t2b1, t2b2, t2b3, (float*)d_out);
}
